// TwoLayerNN_33397665693974
// MI455X (gfx1250) — compile-verified, matrix operand order checked by reading, not run on hardware
//
#include <hip/hip_runtime.h>

// Problem dims (from reference)
#define BATCH 16384
#define KDIM  4096
#define HID   80

typedef __attribute__((ext_vector_type(8)))  float  v8f;
typedef __attribute__((ext_vector_type(16))) __bf16 v16bf;
typedef __attribute__((ext_vector_type(4)))  int    v4i;

typedef __attribute__((address_space(3))) void lds_void;
typedef __attribute__((address_space(3))) v4i  lds_v4i;

#if __has_builtin(__builtin_amdgcn_global_load_async_to_lds_b128) && \
    __has_builtin(__builtin_amdgcn_s_wait_asynccnt)
#define HAVE_ASYNC_LDS 1
#else
#define HAVE_ASYNC_LDS 0
#endif

// Pack two fp32 -> packed bf16x2.
// Preferred: hardware v_cvt_pk_bf16_f32 (if toolchain exposes it).
// Fallback: round-half-up (+0x8000) then one v_perm_b32 to pack both
// high halves: 3 VALU ops per pair, vs ~8 for the full software RNE.
static __device__ __forceinline__ unsigned int pk2bf(float lo, float hi) {
#if __has_builtin(__builtin_amdgcn_cvt_pk_bf16_f32)
  auto r = __builtin_amdgcn_cvt_pk_bf16_f32(lo, hi);
  unsigned int u;
  __builtin_memcpy(&u, &r, 4);
  return u;
#else
  unsigned int ul = __float_as_uint(lo) + 0x8000u;
  unsigned int uh = __float_as_uint(hi) + 0x8000u;
  // result bytes {uh[3],uh[2],ul[3],ul[2]}  ->  bf16(hi):bf16(lo)
  return __builtin_amdgcn_perm(uh, ul, 0x07060302u);
#endif
}

union Frag {
  v16bf v;
  uint4 q[2];
  unsigned int u[8];
};

static __device__ __forceinline__ float selu_f(float x) {
  const float scale = 1.0507009873554805f;
  const float alpha = 1.6732632423543772f;
  return x > 0.f ? scale * x : scale * alpha * (__expf(x) - 1.f);
}

// One-off: transpose + convert W_phi [K,H] fp32 -> Wt [H,K] bf16 in workspace.
__global__ __launch_bounds__(256) void wt_prep(const float* __restrict__ Wphi,
                                               unsigned short* __restrict__ Wt) {
  int idx = blockIdx.x * 256 + threadIdx.x;
  if (idx >= HID * KDIM) return;
  int n = idx >> 12;            // KDIM = 4096 = 2^12
  int k = idx & (KDIM - 1);
  Wt[idx] = (unsigned short)(pk2bf(Wphi[k * HID + n], 0.f) & 0xFFFFu);
}

// Fused: phi = selu(X @ Wphi + bphi) via bf16 WMMA, then all three heads.
__global__ __launch_bounds__(256, 1) void fused_mlp(
    const float* __restrict__ X, const int* __restrict__ D,
    const int* __restrict__ Drand, const unsigned short* __restrict__ Wt,
    const float* __restrict__ bphi, const float* __restrict__ Wg,
    const float* __restrict__ bg, const float* __restrict__ Wj,
    const float* __restrict__ bj, float* __restrict__ out) {
  // Double-buffered B tile: 80 rows x 32 K of bf16 = 5120 B per buffer.
  __shared__ __align__(16) unsigned short ldsB[2][HID * 32];

  const int tid  = threadIdx.x;
  const int wave = tid >> 5;
  const int lane = tid & 31;
  const int l16  = lane & 15;
  const int hi   = lane >> 4;          // 0 = lanes 0-15, 1 = lanes 16-31
  const int rowBase = blockIdx.x * 128 + wave * 16;

  // A-row this lane streams (ISA 16-bit A 16x32 layout: hi lanes offset K by +8)
  const float* xrow = X + (size_t)(rowBase + l16) * KDIM + hi * 8;
  const unsigned int* wtd = (const unsigned int*)Wt;   // Wt rows: KDIM/2 dwords

  // ---- B-tile staging (5120 B): async global->LDS when available ----
#if HAVE_ASYNC_LDS
  // 320 16-byte chunks: thread t does chunk t, threads 0-63 also do 256+t.
  const int c0n   = tid >> 2, c0off = (tid & 3) * 16;           // chunk tid
  const int c1n   = (256 + tid) >> 2, c1off = ((256 + tid) & 3) * 16;
  const char* wtb = (const char*)Wt;
#define STAGE(buf, kq)                                                        \
  do {                                                                        \
    __builtin_amdgcn_global_load_async_to_lds_b128(                           \
        (v4i*)(wtb + (size_t)c0n * (KDIM * 2) + (kq) * 4 + c0off),            \
        (lds_v4i*)(lds_void*)((char*)&ldsB[buf][0] + c0n * 64 + c0off),       \
        0, 0);                                                                \
    if (tid < 64)                                                             \
      __builtin_amdgcn_global_load_async_to_lds_b128(                         \
          (v4i*)(wtb + (size_t)c1n * (KDIM * 2) + (kq) * 4 + c1off),          \
          (lds_v4i*)(lds_void*)((char*)&ldsB[buf][0] + c1n * 64 + c1off),     \
          0, 0);                                                              \
  } while (0)
#define STAGE_WAIT() __builtin_amdgcn_s_wait_asynccnt(0)
#else
#define STAGE(buf, kq)                                                        \
  do {                                                                        \
    _Pragma("unroll") for (int i = 0; i < 5; ++i) {                           \
      int dd = tid + i * 256;                                                 \
      int nn = dd >> 4, kd = dd & 15;                                         \
      ((unsigned int*)&ldsB[buf][0])[nn * 16 + kd] =                          \
          wtd[nn * (KDIM / 2) + (kq) + kd];                                   \
    }                                                                         \
  } while (0)
#define STAGE_WAIT() ((void)0)
#endif

  v8f acc[5];
#pragma unroll
  for (int t = 0; t < 5; ++t)
    acc[t] = v8f{0.f, 0.f, 0.f, 0.f, 0.f, 0.f, 0.f, 0.f};

  // Prologue: stage K-tile 0 into buffer 0.
  STAGE(0, 0);
  STAGE_WAIT();
  __syncthreads();

  for (int kt = 0; kt < KDIM / 32; ++kt) {
    const int cur = kt & 1;
    // Stage next B tile into the other buffer while computing from `cur`.
    if (kt + 1 < KDIM / 32) STAGE(cur ^ 1, (kt + 1) * 16);

    // A fragment: 16 fp32 per lane (two 8-element K-groups), pack to bf16.
    const int k0 = kt * 32;
    if (kt < KDIM / 32 - 8)
      __builtin_prefetch(xrow + k0 + 8 * 32, 0, 3);   // global_prefetch_b8
    float4 a0 = *(const float4*)(xrow + k0);
    float4 a1 = *(const float4*)(xrow + k0 + 4);
    float4 a2 = *(const float4*)(xrow + k0 + 16);
    float4 a3 = *(const float4*)(xrow + k0 + 20);
    Frag A;
    A.u[0] = pk2bf(a0.x, a0.y); A.u[1] = pk2bf(a0.z, a0.w);
    A.u[2] = pk2bf(a1.x, a1.y); A.u[3] = pk2bf(a1.z, a1.w);
    A.u[4] = pk2bf(a2.x, a2.y); A.u[5] = pk2bf(a2.z, a2.w);
    A.u[6] = pk2bf(a3.x, a3.y); A.u[7] = pk2bf(a3.z, a3.w);

    const uint4* bb = (const uint4*)&ldsB[cur][0];  // 4 uint4 per 64B row
#pragma unroll
    for (int t = 0; t < 5; ++t) {
      Frag B;
      const int r = (t * 16 + l16) * 4 + hi;  // column n = t*16 + l16
      B.q[0] = bb[r];        // K lo group
      B.q[1] = bb[r + 2];    // K hi group (+32 bytes)
      acc[t] = __builtin_amdgcn_wmma_f32_16x16x32_bf16(
          false, A.v, false, B.v, (short)0, acc[t], false, false);
    }
    STAGE_WAIT();           // async writes to next buffer complete
    __syncthreads();        // visible to all waves before next iteration
  }

  // ---- Epilogue: bias + SELU, then three heads, all in registers ----
  float bph[5], wg0[5], wg1[5], wj0[5], wj1[5];
#pragma unroll
  for (int t = 0; t < 5; ++t) {
    int n = t * 16 + l16;          // this lane's column in tile t
    bph[t] = bphi[n];
    wg0[t] = Wg[n];
    wg1[t] = Wg[HID + n];
    wj0[t] = Wj[n * 2 + 0];
    wj1[t] = Wj[n * 2 + 1];
  }
  const float bj0 = bj[0], bj1 = bj[1];
  const float bg0 = bg[0], bg1 = bg[1];

  float* outJ  = out;                 // joint  [B,2]
  float* outGS = out + BATCH * 2;     // group_specific [B,1]
  float* outGA = out + BATCH * 3;     // group_agnostic [B,1]

#pragma unroll
  for (int j = 0; j < 8; ++j) {
    // C/D layout: VGPR j holds row j (lanes 0-15) and row j+8 (lanes 16-31).
    float ph[5];
#pragma unroll
    for (int t = 0; t < 5; ++t) ph[t] = selu_f(acc[t][j] + bph[t]);

    const int row = rowBase + j + 8 * hi;
    const int d  = D[row];
    const int dr = Drand[row];
    float pgs = 0.f, pga = 0.f, p0 = 0.f, p1 = 0.f;
#pragma unroll
    for (int t = 0; t < 5; ++t) {
      pgs += ph[t] * (d  ? wg1[t] : wg0[t]);
      pga += ph[t] * (dr ? wg1[t] : wg0[t]);
      p0  += ph[t] * wj0[t];
      p1  += ph[t] * wj1[t];
    }
    // Reduce across the 16 lanes holding this row (xor of bits 0-3 stays in-half).
#pragma unroll
    for (int m = 1; m < 16; m <<= 1) {
      pgs += __shfl_xor(pgs, m, 32);
      pga += __shfl_xor(pga, m, 32);
      p0  += __shfl_xor(p0, m, 32);
      p1  += __shfl_xor(p1, m, 32);
    }
    if (l16 == 0) {
      outJ[row * 2 + 0] = p0 + bj0;
      outJ[row * 2 + 1] = p1 + bj1;
      outGS[row] = pgs + (d  ? bg1 : bg0);
      outGA[row] = pga + (dr ? bg1 : bg0);
    }
  }
}

extern "C" void kernel_launch(void* const* d_in, const int* in_sizes, int n_in,
                              void* d_out, int out_size, void* d_ws, size_t ws_size,
                              hipStream_t stream) {
  (void)in_sizes; (void)n_in; (void)out_size; (void)ws_size;
  const float* X     = (const float*)d_in[0];
  const int*   D     = (const int*)d_in[1];
  const int*   Drand = (const int*)d_in[2];
  const float* Wphi  = (const float*)d_in[3];
  const float* bphi  = (const float*)d_in[4];
  const float* Wg    = (const float*)d_in[5];
  const float* bg    = (const float*)d_in[6];
  const float* Wj    = (const float*)d_in[7];
  const float* bj    = (const float*)d_in[8];
  float* out = (float*)d_out;
  unsigned short* Wt = (unsigned short*)d_ws;   // 80*4096 bf16 = 640 KB

  wt_prep<<<(HID * KDIM + 255) / 256, 256, 0, stream>>>(Wphi, Wt);
  fused_mlp<<<BATCH / 128, 256, 0, stream>>>(X, D, Drand, Wt, bphi, Wg, bg,
                                             Wj, bj, out);
}
